// T2T_module_63367947485276
// MI455X (gfx1250) — matrix operand order repaired and checked by reading, compiled, not run on hardware
//
#include <hip/hip_runtime.h>
#include <math.h>

typedef __bf16 bf16;
typedef __attribute__((ext_vector_type(16))) __bf16 v16bf;
typedef __attribute__((ext_vector_type(8)))  __bf16 v8bf;
typedef __attribute__((ext_vector_type(8)))  float  v8f;

#define LN_EPS 1e-5f
#define PEPS   1e-8f

// ---------------------------------------------------------------- epilogues
enum {
  EPI_F32_BIAS = 0,        // f32 out = acc + bias[n]
  EPI_BF16_BIAS,           // bf16 out = acc + bias[n]
  EPI_BF16_GELU,           // bf16 out = gelu(acc + bias[n])        (exact erf)
  EPI_BF16_EXP,            // bf16 out = exp(acc - rowv[m]) * scale (prm_exp)
  EPI_BF16_SCALE,          // bf16 out = acc / (rowv[m] + eps)      (attention)
  EPI_F32_BIAS_RES_BF16,   // f32 out = acc + bias + (f32)resid_bf16[m,n]
  EPI_F32_BIAS_RES_F32     // f32 out = acc + bias + resid_f32[m,n]
};

// ---------------------------------------------------------------- WMMA GEMM
// C[M,N] = A[M,K](bf16,row-major,lda) * W[N,K](bf16,row-major,ldw)^T + epilogue
// Block = 128 threads = 4 waves stacked along M (block tile 64 x 16*NT).
// Each wave owns a 16 x (16*NT) strip: one A fragment is reused across NT
// B fragments / WMMAs per k-step (raises compute:load from 1:2 to NT:(NT+1)).
// W may be batch-strided (kptv): batch = m0 / tokPerBatch (block tile never
// straddles a batch: 64 | tokens-per-batch).
// M % 64 == 0, N % (16*NT) == 0, K % 32 == 0 (guaranteed by caller).
template<int EPI, int NT>
__global__ __launch_bounds__(128) void gemm_wmma(
    const bf16* __restrict__ A, int lda,
    const bf16* __restrict__ W, int ldw, long wBatchStride, int tokPerBatch,
    const float* __restrict__ bias, const float* __restrict__ rowv,
    const void* __restrict__ resid, int ldr,
    void* __restrict__ Cout, int ldc,
    int M, int N, int K, float scale)
{
  const int lane = threadIdx.x & 31;
  const int w    = threadIdx.x >> 5;
  const int m0   = (blockIdx.y * 4 + w) * 16;
  const int n0   = blockIdx.x * (16 * NT);
  if (m0 >= M) return;   // wave-uniform

  const bf16* Wp = W + (wBatchStride ? (long)(m0 / tokPerBatch) * wBatchStride : 0);

  // per-lane fragment addressing (CDNA5 16-bit A 16x32 / B 32x16 layouts)
  const int arow = m0 + (lane & 15);
  const int kadd = (lane >> 4) << 3;        // A: 0 or 8
  const int bcol = n0 + (lane & 15);
  const int kbdd = (lane >> 4) << 3;        // B: 0 or 8

  v8f acc[NT];
#pragma unroll
  for (int j = 0; j < NT; ++j) acc[j] = (v8f){};

  for (int k0 = 0; k0 < K; k0 += 32) {
    const bf16* pa = A + (long)arow * lda + (k0 + kadd);
    __builtin_prefetch(pa + 32, 0, 1);      // global_prefetch_b8 next k-step
    v8bf alo = *(const v8bf*)pa;
    v8bf ahi = *(const v8bf*)(pa + 16);
    v16bf a;
#pragma unroll
    for (int i = 0; i < 8; ++i) { a[i] = alo[i]; a[i + 8] = ahi[i]; }
#pragma unroll
    for (int j = 0; j < NT; ++j) {
      const bf16* pb = Wp + (long)(bcol + 16 * j) * ldw + (k0 + kbdd);
      v8bf blo = *(const v8bf*)pb;
      v8bf bhi = *(const v8bf*)(pb + 16);
      v16bf b;
#pragma unroll
      for (int i = 0; i < 8; ++i) { b[i] = blo[i]; b[i + 8] = bhi[i]; }
      acc[j] = __builtin_amdgcn_wmma_f32_16x16x32_bf16(
          false, a, false, b, (short)0, acc[j], false, false);
    }
  }

  const int mb = m0 + ((lane >> 4) << 3);
#pragma unroll
  for (int j = 0; j < NT; ++j) {
    const int n = n0 + 16 * j + (lane & 15);
#pragma unroll
    for (int r = 0; r < 8; ++r) {
      const int m = mb + r;
      float v = acc[j][r];
      if (EPI == EPI_F32_BIAS || EPI == EPI_BF16_BIAS || EPI == EPI_BF16_GELU ||
          EPI == EPI_F32_BIAS_RES_BF16 || EPI == EPI_F32_BIAS_RES_F32)
        v += bias[n];
      if (EPI == EPI_BF16_GELU)  v = 0.5f * v * (1.0f + erff(v * 0.70710678118654752f));
      if (EPI == EPI_BF16_EXP)   v = expf(v - rowv[m]) * scale;
      if (EPI == EPI_BF16_SCALE) v = v / (rowv[m] + PEPS);
      if (EPI == EPI_F32_BIAS_RES_BF16)
        v += (float)((const bf16*)resid)[(long)m * ldr + n];
      if (EPI == EPI_F32_BIAS_RES_F32)
        v += ((const float*)resid)[(long)m * ldr + n];
      if (EPI == EPI_F32_BIAS || EPI == EPI_F32_BIAS_RES_BF16 || EPI == EPI_F32_BIAS_RES_F32)
        ((float*)Cout)[(long)m * ldc + n] = v;
      else
        ((bf16*)Cout)[(long)m * ldc + n] = (bf16)v;
    }
  }
}

// ---------------------------------------------------------------- kptv = v^T * kp
// kptv[b, n(emb 64), m(feat 32)] += sum_t v[t,n] * kp[t,m]; WMMA over LDS-transposed
// 128-token chunks (chunk never crosses batch: 128 | tokens-per-batch).
__global__ __launch_bounds__(256) void kptv_wmma(
    const bf16* __restrict__ V, int ldv,       // [T,64] slice (ld=192)
    const bf16* __restrict__ KP,               // [T,32]
    float* __restrict__ kptv,                  // [B,64,32] f32 (pre-zeroed)
    int tokPerBatch)
{
  __shared__ bf16 vT[64 * 128];
  __shared__ bf16 kT[32 * 128];
  const int base = blockIdx.x * 128;
  const int b    = base / tokPerBatch;

  for (int i = threadIdx.x; i < 128 * 64; i += 256) {
    int t = i >> 6, c = i & 63;
    vT[c * 128 + t] = V[(long)(base + t) * ldv + c];
  }
  for (int i = threadIdx.x; i < 128 * 32; i += 256) {
    int t = i >> 5, c = i & 31;
    kT[c * 128 + t] = KP[(long)(base + t) * 32 + c];
  }
  __syncthreads();

  const int lane = threadIdx.x & 31;
  const int w    = threadIdx.x >> 5;          // 8 waves -> 4x2 tiles of 16x16
  const int m0   = (w >> 1) * 16;             // emb 0..63
  const int n0   = (w & 1) * 16;              // feat 0..31
  const int arow = m0 + (lane & 15);
  const int kadd = (lane >> 4) << 3;
  const int bcol = n0 + (lane & 15);
  const int kbdd = (lane >> 4) << 3;

  v8f acc = {};
  for (int k0 = 0; k0 < 128; k0 += 32) {
    const bf16* pa = &vT[arow * 128 + k0 + kadd];
    const bf16* pb = &kT[bcol * 128 + k0 + kbdd];
    v8bf alo = *(const v8bf*)pa;
    v8bf ahi = *(const v8bf*)(pa + 16);
    v8bf blo = *(const v8bf*)pb;
    v8bf bhi = *(const v8bf*)(pb + 16);
    v16bf a, bb;
#pragma unroll
    for (int i = 0; i < 8; ++i) {
      a[i] = alo[i]; a[i + 8] = ahi[i];
      bb[i] = blo[i]; bb[i + 8] = bhi[i];
    }
    acc = __builtin_amdgcn_wmma_f32_16x16x32_bf16(
        false, a, false, bb, (short)0, acc, false, false);
  }
  const int n  = n0 + (lane & 15);
  const int mb = m0 + ((lane >> 4) << 3);
#pragma unroll
  for (int r = 0; r < 8; ++r)
    atomicAdd(&kptv[(long)b * 2048 + (mb + r) * 32 + n], acc[r]);
}

// ---------------------------------------------------------------- elementwise
// stage-1 unfold (k=3,s=1,p=1) of [8,3,224,224] + LN(27) -> bf16 [T,32] (padded)
__global__ __launch_bounds__(128) void unfold1_ln(
    const float* __restrict__ x, const float* __restrict__ g,
    const float* __restrict__ bt, bf16* __restrict__ out)
{
  const int t = blockIdx.x * 128 + threadIdx.x;      // < 401408
  const int b = t / 50176, p = t % 50176;
  const int yy = p / 224, xx = p % 224;
  float v[27]; float s = 0.f;
#pragma unroll
  for (int c = 0; c < 3; ++c)
#pragma unroll
    for (int kh = 0; kh < 3; ++kh)
#pragma unroll
      for (int kw = 0; kw < 3; ++kw) {
        int iy = yy + kh - 1, ix = xx + kw - 1;
        float val = 0.f;
        if (iy >= 0 && iy < 224 && ix >= 0 && ix < 224)
          val = x[(((long)(b * 3 + c) * 224 + iy) * 224) + ix];
        v[c * 9 + kh * 3 + kw] = val; s += val;
      }
  const float mu = s * (1.f / 27.f);
  float var = 0.f;
#pragma unroll
  for (int i = 0; i < 27; ++i) { float d = v[i] - mu; var += d * d; }
  const float rs = rsqrtf(var * (1.f / 27.f) + LN_EPS);
  bf16* o = out + (long)t * 32;
#pragma unroll
  for (int i = 0; i < 27; ++i) o[i] = (bf16)((v[i] - mu) * rs * g[i] + bt[i]);
#pragma unroll
  for (int i = 27; i < 32; ++i) o[i] = (bf16)0.f;
}

// unfold (k=3,s=2,p=1) of token image [B,S,S,64] -> 576 features (+optional LN)
template<bool DO_LN>
__global__ __launch_bounds__(192) void unfold576(
    const float* __restrict__ tin, int S,
    const float* __restrict__ g, const float* __restrict__ bt,
    bf16* __restrict__ out)
{
  const int So = S >> 1;
  const int token = blockIdx.x;
  const int b = token / (So * So), p = token % (So * So);
  const int oy = p / So, ox = p % So;
  const int tid = threadIdx.x;
  __shared__ float red[192];

  float vals[3]; float s = 0.f;
#pragma unroll
  for (int j = 0; j < 3; ++j) {
    const int f = tid + j * 192;
    const int c = f / 9, k = f % 9;
    const int kh = k / 3, kw = k % 3;
    const int iy = 2 * oy + kh - 1, ix = 2 * ox + kw - 1;
    float val = 0.f;
    if (iy >= 0 && iy < S && ix >= 0 && ix < S)
      val = tin[((long)(b * S + iy) * S + ix) * 64 + c];
    vals[j] = val; s += val;
  }
  if (DO_LN) {
    red[tid] = s; __syncthreads();
    if (tid < 96) red[tid] += red[tid + 96]; __syncthreads();
    if (tid < 48) red[tid] += red[tid + 48]; __syncthreads();
    if (tid < 24) red[tid] += red[tid + 24]; __syncthreads();
    if (tid < 12) red[tid] += red[tid + 12]; __syncthreads();
    if (tid == 0) {
      float t0 = 0.f;
      for (int i = 0; i < 12; ++i) t0 += red[i];
      red[0] = t0;
    }
    __syncthreads();
    const float mu = red[0] * (1.f / 576.f);
    __syncthreads();
    float vs = 0.f;
#pragma unroll
    for (int j = 0; j < 3; ++j) { float d = vals[j] - mu; vs += d * d; }
    red[tid] = vs; __syncthreads();
    if (tid < 96) red[tid] += red[tid + 96]; __syncthreads();
    if (tid < 48) red[tid] += red[tid + 48]; __syncthreads();
    if (tid < 24) red[tid] += red[tid + 24]; __syncthreads();
    if (tid < 12) red[tid] += red[tid + 12]; __syncthreads();
    if (tid == 0) {
      float t0 = 0.f;
      for (int i = 0; i < 12; ++i) t0 += red[i];
      red[0] = t0;
    }
    __syncthreads();
    const float rs = rsqrtf(red[0] * (1.f / 576.f) + LN_EPS);
#pragma unroll
    for (int j = 0; j < 3; ++j) {
      const int f = tid + j * 192;
      out[(long)token * 576 + f] = (bf16)((vals[j] - mu) * rs * g[f] + bt[f]);
    }
  } else {
#pragma unroll
    for (int j = 0; j < 3; ++j)
      out[(long)token * 576 + tid + j * 192] = (bf16)vals[j];
  }
}

// xd[t] = 0.5 * sum_i A[t,i]^2   (emb = 64, bf16 slice)
__global__ void rowsq64(const bf16* __restrict__ A, int lda,
                        float* __restrict__ xd, int T)
{
  const int t = blockIdx.x * 256 + threadIdx.x;
  if (t >= T) return;
  const bf16* p = A + (long)t * lda;
  float s = 0.f;
#pragma unroll
  for (int i = 0; i < 64; ++i) { float v = (float)p[i]; s += v * v; }
  xd[t] = 0.5f * s;
}

// kpsum[b,m] += sum over 256-token chunk (256 | tokens-per-batch)
__global__ __launch_bounds__(256) void colsum32(
    const bf16* __restrict__ KP, float* __restrict__ kpsum, int tokPerBatch)
{
  __shared__ float red[256];
  const int base = blockIdx.x * 256;
  const int b = base / tokPerBatch;
  const int m = threadIdx.x & 31, grp = threadIdx.x >> 5;
  float s = 0.f;
#pragma unroll
  for (int j = 0; j < 32; ++j)
    s += (float)KP[(long)(base + grp + j * 8) * 32 + m];
  red[threadIdx.x] = s; __syncthreads();
  if (threadIdx.x < 128) red[threadIdx.x] += red[threadIdx.x + 128]; __syncthreads();
  if (threadIdx.x < 64)  red[threadIdx.x] += red[threadIdx.x + 64];  __syncthreads();
  if (threadIdx.x < 32)
    atomicAdd(&kpsum[b * 32 + m], red[threadIdx.x] + red[threadIdx.x + 32]);
}

// D[t] = qp[t,:] . kpsum[b,:]
__global__ void rowdot32(const bf16* __restrict__ QP, const float* __restrict__ kpsum,
                         float* __restrict__ D, int T, int tokPerBatch)
{
  const int t = blockIdx.x * 256 + threadIdx.x;
  if (t >= T) return;
  const int b = t / tokPerBatch;
  float s = 0.f;
#pragma unroll
  for (int m = 0; m < 32; ++m)
    s += (float)QP[(long)t * 32 + m] * kpsum[b * 32 + m];
  D[t] = s;
}

// LN over 64 f32 features -> bf16
__global__ void ln64(const float* __restrict__ y, const float* __restrict__ g,
                     const float* __restrict__ bt, bf16* __restrict__ out, int T)
{
  const int t = blockIdx.x * 256 + threadIdx.x;
  if (t >= T) return;
  const float4* p = (const float4*)(y + (long)t * 64);
  float buf[64]; float s = 0.f;
#pragma unroll
  for (int i = 0; i < 16; ++i) {
    float4 q = p[i];
    buf[4 * i] = q.x; buf[4 * i + 1] = q.y; buf[4 * i + 2] = q.z; buf[4 * i + 3] = q.w;
    s += q.x + q.y + q.z + q.w;
  }
  const float mu = s * (1.f / 64.f);
  float var = 0.f;
#pragma unroll
  for (int i = 0; i < 64; ++i) { float d = buf[i] - mu; var += d * d; }
  const float rs = rsqrtf(var * (1.f / 64.f) + LN_EPS);
  bf16* o = out + (long)t * 64;
#pragma unroll
  for (int i = 0; i < 64; ++i) o[i] = (bf16)((buf[i] - mu) * rs * g[i] + bt[i]);
}

// f32 [N,K] -> bf16 [N,Kpad], zero-padded K..Kpad
__global__ void cvt_pad_bf16(const float* __restrict__ src, bf16* __restrict__ dst,
                             int N, int K, int Kpad)
{
  const long i = (long)blockIdx.x * 256 + threadIdx.x;
  if (i >= (long)N * Kpad) return;
  const int n = (int)(i / Kpad), k = (int)(i % Kpad);
  dst[i] = (k < K) ? (bf16)src[(long)n * K + k] : (bf16)0.f;
}

// ================================================================ host side
namespace {
constexpr int Bn = 8;
constexpr int T1 = 8 * 224 * 224;   // 401408
constexpr int T2 = 8 * 112 * 112;   // 100352
constexpr int T3 = 8 * 56 * 56;     // 25088
constexpr float INV_SQRT_M = 0.17677669529663689f;  // 1/sqrt(32)

struct Scratch {
  bf16 *kqv, *kp, *qp, *att, *h2, *hid, *kptv_bf;
  float *xd, *Dval, *y, *kpsum, *kptv;
};

// block tile = 64(M) x 16*NT(N)
template<int NT> inline dim3 ggrid(int M, int N) { return dim3(N / (16 * NT), M / 64); }

void run_performer(hipStream_t s, const bf16* tok, int Kdim, int T, int tpb,
                   const bf16* wkqv, const float* bkqv,
                   const bf16* wproj, const float* bproj,
                   const float* g2, const float* b2,
                   const bf16* wm1, const float* bm1,
                   const bf16* wm2, const float* bm2,
                   const bf16* wfeat, const Scratch& sc, float* tout)
{
  // kqv = LN(tok) @ Wkqv^T + b   -> bf16 [T,192]
  gemm_wmma<EPI_BF16_BIAS, 4><<<ggrid<4>(T, 192), 128, 0, s>>>(
      tok, Kdim, wkqv, Kdim, 0, 1, bkqv, nullptr, nullptr, 0,
      sc.kqv, 192, T, 192, Kdim, 0.f);
  // kp = prm_exp(k)
  rowsq64<<<(T + 255) / 256, 256, 0, s>>>(sc.kqv, 192, sc.xd, T);
  gemm_wmma<EPI_BF16_EXP, 2><<<ggrid<2>(T, 32), 128, 0, s>>>(
      sc.kqv, 192, wfeat, 64, 0, 1, nullptr, sc.xd, nullptr, 0,
      sc.kp, 32, T, 32, 64, INV_SQRT_M);
  // qp = prm_exp(q)
  rowsq64<<<(T + 255) / 256, 256, 0, s>>>(sc.kqv + 64, 192, sc.xd, T);
  gemm_wmma<EPI_BF16_EXP, 2><<<ggrid<2>(T, 32), 128, 0, s>>>(
      sc.kqv + 64, 192, wfeat, 64, 0, 1, nullptr, sc.xd, nullptr, 0,
      sc.qp, 32, T, 32, 64, INV_SQRT_M);
  // kpsum, kptv
  hipMemsetAsync(sc.kpsum, 0, Bn * 32 * sizeof(float), s);
  hipMemsetAsync(sc.kptv, 0, Bn * 2048 * sizeof(float), s);
  colsum32<<<T / 256, 256, 0, s>>>(sc.kp, sc.kpsum, tpb);
  kptv_wmma<<<T / 128, 256, 0, s>>>(sc.kqv + 128, 192, sc.kp, sc.kptv, tpb);
  cvt_pad_bf16<<<(Bn * 2048 + 255) / 256, 256, 0, s>>>(
      sc.kptv, sc.kptv_bf, Bn * 64, 32, 32);
  // D = qp . kpsum ; att = (qp @ kptv^T) / (D + eps)
  rowdot32<<<(T + 255) / 256, 256, 0, s>>>(sc.qp, sc.kpsum, sc.Dval, T, tpb);
  gemm_wmma<EPI_BF16_SCALE, 4><<<ggrid<4>(T, 64), 128, 0, s>>>(
      sc.qp, 32, sc.kptv_bf, 32, 2048, tpb, nullptr, sc.Dval, nullptr, 0,
      sc.att, 64, T, 64, 32, 0.f);
  // y = v + att @ Wproj^T + b
  gemm_wmma<EPI_F32_BIAS_RES_BF16, 4><<<ggrid<4>(T, 64), 128, 0, s>>>(
      sc.att, 64, wproj, 64, 0, 1, bproj, nullptr, sc.kqv + 128, 192,
      sc.y, 64, T, 64, 64, 0.f);
  // tout = y + m2( gelu( m1( LN(y) ) ) )
  ln64<<<(T + 255) / 256, 256, 0, s>>>(sc.y, g2, b2, sc.h2, T);
  gemm_wmma<EPI_BF16_GELU, 4><<<ggrid<4>(T, 64), 128, 0, s>>>(
      sc.h2, 64, wm1, 64, 0, 1, bm1, nullptr, nullptr, 0,
      sc.hid, 64, T, 64, 64, 0.f);
  gemm_wmma<EPI_F32_BIAS_RES_F32, 4><<<ggrid<4>(T, 64), 128, 0, s>>>(
      sc.hid, 64, wm2, 64, 0, 1, bm2, nullptr, sc.y, 64,
      tout, 64, T, 64, 64, 0.f);
}
} // namespace

extern "C" void kernel_launch(void* const* d_in, const int* in_sizes, int n_in,
                              void* d_out, int out_size, void* d_ws, size_t ws_size,
                              hipStream_t stream) {
  (void)in_sizes; (void)n_in; (void)out_size; (void)ws_size;
  const float* x = (const float*)d_in[0];
  // p1: 1..13 ; p2: 14..26 ; proj: 27,28
  const float *p1_kqv_w = (const float*)d_in[1],  *p1_kqv_b = (const float*)d_in[2];
  const float *p1_proj_w = (const float*)d_in[3], *p1_proj_b = (const float*)d_in[4];
  const float *p1_n1_g = (const float*)d_in[5],   *p1_n1_b = (const float*)d_in[6];
  const float *p1_n2_g = (const float*)d_in[7],   *p1_n2_b = (const float*)d_in[8];
  const float *p1_m1_w = (const float*)d_in[9],   *p1_m1_b = (const float*)d_in[10];
  const float *p1_m2_w = (const float*)d_in[11],  *p1_m2_b = (const float*)d_in[12];
  const float *p1_w = (const float*)d_in[13];
  const float *p2_kqv_w = (const float*)d_in[14], *p2_kqv_b = (const float*)d_in[15];
  const float *p2_proj_w = (const float*)d_in[16],*p2_proj_b = (const float*)d_in[17];
  const float *p2_n1_g = (const float*)d_in[18],  *p2_n1_b = (const float*)d_in[19];
  const float *p2_n2_g = (const float*)d_in[20],  *p2_n2_b = (const float*)d_in[21];
  const float *p2_m1_w = (const float*)d_in[22],  *p2_m1_b = (const float*)d_in[23];
  const float *p2_m2_w = (const float*)d_in[24],  *p2_m2_b = (const float*)d_in[25];
  const float *p2_w = (const float*)d_in[26];
  const float *proj_w = (const float*)d_in[27],   *proj_b = (const float*)d_in[28];

  char* ws = (char*)d_ws;
  size_t off = 0;
  auto alloc = [&](size_t bytes) -> char* {
    char* p = ws + off;
    off = (off + bytes + 255) & ~(size_t)255;
    return p;
  };

  // bf16 weights
  bf16* wkqv1 = (bf16*)alloc(192 * 32 * 2);
  bf16* wproj1 = (bf16*)alloc(64 * 64 * 2);
  bf16* wm11 = (bf16*)alloc(64 * 64 * 2);
  bf16* wm21 = (bf16*)alloc(64 * 64 * 2);
  bf16* wfeat1 = (bf16*)alloc(32 * 64 * 2);
  bf16* wkqv2 = (bf16*)alloc(192 * 576 * 2);
  bf16* wproj2 = (bf16*)alloc(64 * 64 * 2);
  bf16* wm12 = (bf16*)alloc(64 * 64 * 2);
  bf16* wm22 = (bf16*)alloc(64 * 64 * 2);
  bf16* wfeat2 = (bf16*)alloc(32 * 64 * 2);
  bf16* wprojF = (bf16*)alloc(768 * 576 * 2);

  // activations (scratch sized for stage 1, reused by stage 2)
  bf16* tok1 = (bf16*)alloc((size_t)T1 * 32 * 2);
  Scratch sc;
  sc.kqv = (bf16*)alloc((size_t)T1 * 192 * 2);
  sc.kp  = (bf16*)alloc((size_t)T1 * 32 * 2);
  sc.qp  = (bf16*)alloc((size_t)T1 * 32 * 2);
  sc.att = (bf16*)alloc((size_t)T1 * 64 * 2);
  sc.h2  = (bf16*)alloc((size_t)T1 * 64 * 2);
  sc.hid = (bf16*)alloc((size_t)T1 * 64 * 2);
  sc.kptv_bf = (bf16*)alloc(Bn * 2048 * 2);
  sc.xd   = (float*)alloc((size_t)T1 * 4);
  sc.Dval = (float*)alloc((size_t)T1 * 4);
  sc.y    = (float*)alloc((size_t)T1 * 64 * 4);
  sc.kpsum = (float*)alloc(Bn * 32 * 4);
  sc.kptv  = (float*)alloc(Bn * 2048 * 4);
  float* tout1 = (float*)alloc((size_t)T1 * 64 * 4);
  bf16*  tok2  = (bf16*)alloc((size_t)T2 * 576 * 2);
  float* tout2 = (float*)alloc((size_t)T2 * 64 * 4);
  bf16*  tok3  = (bf16*)alloc((size_t)T3 * 576 * 2);

  // ---- weight conversion
  auto cvt = [&](const float* s, bf16* d, int N, int K, int Kp) {
    cvt_pad_bf16<<<(int)(((long)N * Kp + 255) / 256), 256, 0, stream>>>(s, d, N, K, Kp);
  };
  cvt(p1_kqv_w, wkqv1, 192, 27, 32);
  cvt(p1_proj_w, wproj1, 64, 64, 64);
  cvt(p1_m1_w, wm11, 64, 64, 64);
  cvt(p1_m2_w, wm21, 64, 64, 64);
  cvt(p1_w, wfeat1, 32, 64, 64);
  cvt(p2_kqv_w, wkqv2, 192, 576, 576);
  cvt(p2_proj_w, wproj2, 64, 64, 64);
  cvt(p2_m1_w, wm12, 64, 64, 64);
  cvt(p2_m2_w, wm22, 64, 64, 64);
  cvt(p2_w, wfeat2, 32, 64, 64);
  cvt(proj_w, wprojF, 768, 576, 576);

  // ---- stage 1
  unfold1_ln<<<T1 / 128, 128, 0, stream>>>(x, p1_n1_g, p1_n1_b, tok1);
  run_performer(stream, tok1, 32, T1, 50176,
                wkqv1, p1_kqv_b, wproj1, p1_proj_b, p1_n2_g, p1_n2_b,
                wm11, p1_m1_b, wm21, p1_m2_b, wfeat1, sc, tout1);

  // ---- stage 2
  unfold576<true><<<T2, 192, 0, stream>>>(tout1, 224, p2_n1_g, p2_n1_b, tok2);
  run_performer(stream, tok2, 576, T2, 12544,
                wkqv2, p2_kqv_b, wproj2, p2_proj_b, p2_n2_g, p2_n2_b,
                wm12, p2_m1_b, wm22, p2_m2_b, wfeat2, sc, tout2);

  // ---- stage 3: unfold + final projection [T3,576] x [576,768]
  unfold576<false><<<T3, 192, 0, stream>>>(tout2, 112, nullptr, nullptr, tok3);
  gemm_wmma<EPI_F32_BIAS, 4><<<ggrid<4>(T3, 768), 128, 0, stream>>>(
      tok3, 576, wprojF, 576, 0, 1, proj_b, nullptr, nullptr, 0,
      (float*)d_out, 768, T3, 768, 576, 0.f);
}
